// MHA_463856468092
// MI455X (gfx1250) — matrix operand order repaired and checked by reading, not compiled, not run on hardware
//
#include <hip/hip_runtime.h>

#define Bc 2
#define Sc 2048
#define Ec 2048
#define Hc 16
#define Dc 128

typedef __bf16 bf16_t;
typedef __attribute__((ext_vector_type(16))) __bf16 v16bf;
typedef __attribute__((ext_vector_type(8)))  __bf16 v8bf;
typedef __attribute__((ext_vector_type(4)))  __bf16 v4bf;
typedef __attribute__((ext_vector_type(8)))  float  v8f;
typedef unsigned int u32x4 __attribute__((ext_vector_type(4)));
typedef int i32x8 __attribute__((ext_vector_type(8)));
typedef int i32x4 __attribute__((ext_vector_type(4)));

static __device__ __forceinline__ v16bf cat16(v8bf lo, v8bf hi) {
    union { v16bf v; v8bf h[2]; } u;
    u.h[0] = lo; u.h[1] = hi;
    return u.v;
}

static __device__ __forceinline__ v8f wmma_bf16(v16bf a, v16bf b, v8f c) {
    // D = A(16x32 bf16) * B(32x16 bf16) + C(16x16 f32)
    return __builtin_amdgcn_wmma_f32_16x16x32_bf16(
        false, a, false, b, (short)0, c, false, false);
}

// LDS byte offset of a __shared__ pointer (generic -> AS(3) -> int)
static __device__ __forceinline__ unsigned lds_off_u32(const void* p) {
    return (unsigned)(unsigned long long)(uintptr_t)
        (__attribute__((address_space(3))) const void*)p;
}

// ---------------------------------------------------------------------------
// Tensor Data Mover: async 2D bf16 tile load global->LDS with HW row padding.
// D# per CDNA5 ISA 8.3/8.4. One issue per workgroup (TDM ignores EXEC; gate
// by wave id). Completion via TENSORcnt.
//   padIntervalEnc: pad after 2^(enc+1) dwords;  padAmountEnc: (enc+1) dwords
// ---------------------------------------------------------------------------
static __device__ __forceinline__ void tdm_load_2d_bf16(
    unsigned ldsOff, const bf16_t* g, unsigned cols, unsigned rows,
    unsigned strideElems, unsigned padIntervalEnc, unsigned padAmountEnc) {
    unsigned long long ga = (unsigned long long)(uintptr_t)g;
    u32x4 g0;
    g0[0] = 1u;                                   // count=1 (valid descriptor)
    g0[1] = ldsOff;                               // lds_addr
    g0[2] = (unsigned)ga;                         // global_addr[31:0]
    g0[3] = (unsigned)((ga >> 32) & 0x01FFFFFFu)  // global_addr[56:32]
          | (2u << 30);                           // type=2 ("image")
    i32x8 g1;
    g1[0] = (int)((1u << 16)                      // data_size = 2 bytes
                | (1u << 20)                      // pad_enable
                | (padIntervalEnc << 22)
                | (padAmountEnc << 25));
    g1[1] = (int)((cols & 0xFFFFu) << 16);        // tensor_dim0[15:0]
    g1[2] = (int)(((cols >> 16) & 0xFFFFu)        // tensor_dim0[31:16]
                | ((rows & 0xFFFFu) << 16));      // tensor_dim1[15:0]
    g1[3] = (int)(((rows >> 16) & 0xFFFFu)        // tensor_dim1[31:16]
                | ((cols & 0xFFFFu) << 16));      // tile_dim0
    g1[4] = (int)(rows & 0xFFFFu);                // tile_dim1 (tile_dim2=0)
    g1[5] = (int)strideElems;                     // tensor_dim0_stride[31:0]
    g1[6] = 0;                                    // stride0 hi | stride1 lo
    g1[7] = 0;
    i32x4 z4 = {0, 0, 0, 0};
    i32x8 z8 = {0, 0, 0, 0, 0, 0, 0, 0};
    // clang-23 / therock signature: (g0, g1, g2, g3, extra, cpol)
    __builtin_amdgcn_tensor_load_to_lds(g0, g1, z4, z4, z8, 0);
}

// ---------------------------------------------------------------------------
// fp32 -> bf16 conversion (vectorized x4)
// ---------------------------------------------------------------------------
__global__ void cvt_f32_bf16(const float* __restrict__ in,
                             bf16_t* __restrict__ out, int n4) {
    int i = blockIdx.x * blockDim.x + threadIdx.x;
    if (i < n4) {
        float4 f = ((const float4*)in)[i];
        v4bf o = { (bf16_t)f.x, (bf16_t)f.y, (bf16_t)f.z, (bf16_t)f.w };
        *(v4bf*)(out + 4 * (size_t)i) = o;
    }
}

// ---------------------------------------------------------------------------
// Tiled WMMA GEMM: C[M,N] = A[M,K] * W[N,K]^T + bias[N]
// 256 threads = 8 wave32 waves; tile 128x128, K-step 32; TDM-staged,
// double-buffered LDS (DMA of tile k+1 overlaps WMMA on tile k).
// ---------------------------------------------------------------------------
template <bool QKV>
__global__ __launch_bounds__(256) void gemm_wmma(
    const bf16_t* __restrict__ A, const bf16_t* __restrict__ W,
    const float* __restrict__ bias,
    bf16_t* __restrict__ qO, bf16_t* __restrict__ kO, bf16_t* __restrict__ vO,
    float* __restrict__ outF, int M, int N, int K) {
    constexpr int BM = 128, BN = 128, BK = 32;
    constexpr int PA = 40, PB = 40;  // LDS pitch = 32 + 8 bf16 (TDM pad)
    __shared__ __align__(16) bf16_t As[2][BM * PA];
    __shared__ __align__(16) bf16_t Bs[2][BN * PB];

    const int tid  = threadIdx.x;
    const int lane = tid & 31, wid = tid >> 5;
    const int half = (lane >> 4) & 1, l16 = lane & 15;
    const int waveM = wid & 3, waveN = wid >> 2;
    const int m0 = blockIdx.y * BM, n0 = blockIdx.x * BN;
    const bool leader = (wid == 0);

    const bf16_t* gA = A + (size_t)m0 * K;
    const bf16_t* gW = W + (size_t)n0 * K;

    const v8f vzero = {0.f, 0.f, 0.f, 0.f, 0.f, 0.f, 0.f, 0.f};
    v8f acc[2][4];
#pragma unroll
    for (int i = 0; i < 2; ++i)
#pragma unroll
        for (int j = 0; j < 4; ++j) acc[i][j] = vzero;

    if (leader) {
        tdm_load_2d_bf16(lds_off_u32(As[0]), gA, BK, BM, K, 3, 3);
        tdm_load_2d_bf16(lds_off_u32(Bs[0]), gW, BK, BN, K, 3, 3);
        __builtin_amdgcn_s_wait_tensorcnt(0);
    }
    __syncthreads();

    for (int k0 = 0; k0 < K; k0 += BK) {
        const int  buf = (k0 >> 5) & 1;
        const bool hasNext = (k0 + BK) < K;
        if (hasNext && leader) {
            tdm_load_2d_bf16(lds_off_u32(As[buf ^ 1]), gA + k0 + BK, BK, BM, K, 3, 3);
            tdm_load_2d_bf16(lds_off_u32(Bs[buf ^ 1]), gW + k0 + BK, BK, BN, K, 3, 3);
        }

        v16bf af[2], bfv[4];
#pragma unroll
        for (int mt = 0; mt < 2; ++mt) {
            const bf16_t* p = As[buf] + (waveM * 32 + mt * 16 + l16) * PA + half * 8;
            af[mt] = cat16(*(const v8bf*)p, *(const v8bf*)(p + 16));
        }
#pragma unroll
        for (int nt = 0; nt < 4; ++nt) {
            const bf16_t* p = Bs[buf] + (waveN * 64 + nt * 16 + l16) * PB + half * 8;
            bfv[nt] = cat16(*(const v8bf*)p, *(const v8bf*)(p + 16));
        }
#pragma unroll
        for (int mt = 0; mt < 2; ++mt)
#pragma unroll
            for (int nt = 0; nt < 4; ++nt)
                acc[mt][nt] = wmma_bf16(af[mt], bfv[nt], acc[mt][nt]);

        __syncthreads();                       // all waves done reading buf
        if (hasNext) {
            if (leader) __builtin_amdgcn_s_wait_tensorcnt(0);
            __syncthreads();                   // publish buf^1
        }
    }

    // Epilogue. C layout: lane owns (m = r + half*8, n = l16) for VGPR r.
#pragma unroll
    for (int mt = 0; mt < 2; ++mt) {
#pragma unroll
        for (int nt = 0; nt < 4; ++nt) {
            const int ng = n0 + waveN * 64 + nt * 16 + l16;
            const float bia = bias[ng];
#pragma unroll
            for (int r = 0; r < 8; ++r) {
                const int mg = m0 + waveM * 32 + mt * 16 + half * 8 + r;
                float val = acc[mt][nt][r] + bia;
                if constexpr (QKV) {
                    const int part = ng / Ec, rem = ng % Ec;
                    const int hh = rem / Dc, dd = rem % Dc;
                    const int bb = mg / Sc, ss = mg % Sc;
                    bf16_t* dst = (part == 0) ? qO : ((part == 1) ? kO : vO);
                    dst[(((size_t)bb * Hc + hh) * Sc + ss) * Dc + dd] = (bf16_t)val;
                } else {
                    outF[(size_t)mg * N + ng] = val;
                }
            }
        }
    }
}

// ---------------------------------------------------------------------------
// Flash attention: grid (S/128, B*H), block 256 = 8 waves; 32-key tiles with
// online softmax. K tiles: TDM double-buffered (HW-padded rows). V tiles:
// register-pipelined manual transpose into LDS (TDM cannot transpose).
// ---------------------------------------------------------------------------
__global__ __launch_bounds__(256) void attn_flash(
    const bf16_t* __restrict__ q, const bf16_t* __restrict__ k,
    const bf16_t* __restrict__ v, bf16_t* __restrict__ ctx) {
    constexpr int KT = 32, PK = 136, PV = 40, PP = 40;
    __shared__ __align__(16) bf16_t Ks[2][KT * PK]; // [key][d], TDM-padded
    __shared__ __align__(16) bf16_t Vt[Dc * PV];    // [d][key]
    __shared__ __align__(16) bf16_t Ps[8][16 * PP]; // per-wave P tile [m][key]

    const int tid  = threadIdx.x;
    const int lane = tid & 31, wid = tid >> 5;
    const int half = (lane >> 4) & 1, l16 = lane & 15;
    const int bh = blockIdx.y;                 // b*H + h
    const int s0 = blockIdx.x * 128;           // q row base for block
    const int bb = bh / Hc, hh = bh % Hc;
    const bool leader = (wid == 0);

    // Resident Q fragments: 16 rows x 128 d = 4 K-steps of 32
    const bf16_t* qb = q + ((size_t)bh * Sc + s0 + wid * 16) * Dc;
    v16bf qf[4];
#pragma unroll
    for (int kk = 0; kk < 4; ++kk) {
        const bf16_t* p = qb + (size_t)l16 * Dc + kk * 32 + half * 8;
        qf[kk] = cat16(*(const v8bf*)p, *(const v8bf*)(p + 16));
    }

    // V staging: each thread owns 2x16B per tile
    const int ldRow = tid >> 4;            // 0..15 (+16 second chunk)
    const int ldCol = (tid & 15) << 3;     // 0..120
    const bf16_t* gK = k + (size_t)bh * Sc * Dc;
    const bf16_t* gV = v + ((size_t)bh * Sc + ldRow) * Dc + ldCol;
    const size_t rowStep = (size_t)16 * Dc;

    const v8f vzero = {0.f, 0.f, 0.f, 0.f, 0.f, 0.f, 0.f, 0.f};
    v8f accc[8];
#pragma unroll
    for (int t = 0; t < 8; ++t) accc[t] = vzero;
    float mrow[8], lrow[8];
#pragma unroll
    for (int r = 0; r < 8; ++r) { mrow[r] = -3.0e38f; lrow[r] = 0.f; }
    // exp(scale*(s - m)) == exp2(CEXP*(s - m)); running max kept in raw units
    const float CEXP = 0.08838834764831845f * 1.4426950408889634f;

    // Prologue: K tile 0 via TDM, V tile 0 via registers
    v8bf rV[2];
    if (leader) tdm_load_2d_bf16(lds_off_u32(Ks[0]), gK, Dc, KT, Dc, 5, 3);
#pragma unroll
    for (int i = 0; i < 2; ++i) rV[i] = *(const v8bf*)(gV + i * rowStep);
#pragma unroll
    for (int i = 0; i < 2; ++i)
#pragma unroll
        for (int j = 0; j < 8; ++j)
            Vt[(ldCol + j) * PV + ldRow + i * 16] = rV[i][j];
    if (leader) __builtin_amdgcn_s_wait_tensorcnt(0);
    __syncthreads();

    for (int kt = 0; kt < Sc; kt += KT) {
        const int  buf = (kt >> 5) & 1;
        const bool hasNext = (kt + KT) < Sc;
        if (hasNext) {
            if (leader)
                tdm_load_2d_bf16(lds_off_u32(Ks[buf ^ 1]),
                                 gK + (size_t)(kt + KT) * Dc, Dc, KT, Dc, 5, 3);
            const size_t off = (size_t)(kt + KT) * Dc;
#pragma unroll
            for (int i = 0; i < 2; ++i) rV[i] = *(const v8bf*)(gV + off + i * rowStep);
        }

        // scores: Q(16x128) * K^T(128x32) -> 2 tiles of 16x16
        v8f sc[2];
        sc[0] = vzero; sc[1] = vzero;
#pragma unroll
        for (int t = 0; t < 2; ++t)
#pragma unroll
            for (int kk = 0; kk < 4; ++kk) {
                const bf16_t* p = Ks[buf] + (t * 16 + l16) * PK + kk * 32 + half * 8;
                v16bf bfr = cat16(*(const v8bf*)p, *(const v8bf*)(p + 16));
                sc[t] = wmma_bf16(qf[kk], bfr, sc[t]);
            }

        // online softmax; row m = r + half*8 owned by one 16-lane half
        bf16_t* prow = Ps[wid];
#pragma unroll
        for (int r = 0; r < 8; ++r) {
            float a0 = sc[0][r], a1 = sc[1][r];
            float mloc = fmaxf(a0, a1);
            mloc = fmaxf(mloc, __shfl_xor(mloc, 8, 32));
            mloc = fmaxf(mloc, __shfl_xor(mloc, 4, 32));
            mloc = fmaxf(mloc, __shfl_xor(mloc, 2, 32));
            mloc = fmaxf(mloc, __shfl_xor(mloc, 1, 32));
            float mnew  = fmaxf(mrow[r], mloc);
            float alpha = exp2f((mrow[r] - mnew) * CEXP);
            float p0 = exp2f((a0 - mnew) * CEXP);
            float p1 = exp2f((a1 - mnew) * CEXP);
            float lsum = p0 + p1;
            lsum += __shfl_xor(lsum, 8, 32);
            lsum += __shfl_xor(lsum, 4, 32);
            lsum += __shfl_xor(lsum, 2, 32);
            lsum += __shfl_xor(lsum, 1, 32);
            lrow[r] = lrow[r] * alpha + lsum;
            mrow[r] = mnew;
#pragma unroll
            for (int t2 = 0; t2 < 8; ++t2) accc[t2][r] *= alpha;
            prow[(half * 8 + r) * PP + l16]      = (bf16_t)p0;
            prow[(half * 8 + r) * PP + 16 + l16] = (bf16_t)p1;
        }
        // wave-local LDS RAW fence (same-wave DS ops; cheaper than barrier)
        asm volatile("s_wait_dscnt 0x0" ::: "memory");

        // P(16x32) * V(32x128): one A fragment, 8 output d-tiles
        {
            const bf16_t* pp = prow + (size_t)l16 * PP + half * 8;
            v16bf pf = cat16(*(const v8bf*)pp, *(const v8bf*)(pp + 16));
#pragma unroll
            for (int t2 = 0; t2 < 8; ++t2) {
                const bf16_t* vp = Vt + (t2 * 16 + l16) * PV + half * 8;
                v16bf vf = cat16(*(const v8bf*)vp, *(const v8bf*)(vp + 16));
                accc[t2] = wmma_bf16(pf, vf, accc[t2]);
            }
        }
        __syncthreads();                       // done reading Ks[buf], Vt
        if (hasNext) {
#pragma unroll
            for (int i = 0; i < 2; ++i)
#pragma unroll
                for (int j = 0; j < 8; ++j)
                    Vt[(ldCol + j) * PV + ldRow + i * 16] = rV[i][j];
            if (leader) __builtin_amdgcn_s_wait_tensorcnt(0);
            __syncthreads();                   // publish Ks[buf^1], new Vt
        }
    }

    // epilogue: ctx[b, s, h*D + d] bf16
    bf16_t* cb = ctx + ((size_t)(bb * Sc) + s0 + wid * 16) * Ec + hh * Dc;
#pragma unroll
    for (int t2 = 0; t2 < 8; ++t2) {
#pragma unroll
        for (int r = 0; r < 8; ++r) {
            float val = accc[t2][r] / lrow[r];
            cb[(size_t)(half * 8 + r) * Ec + t2 * 16 + l16] = (bf16_t)val;
        }
    }
}

// ---------------------------------------------------------------------------
extern "C" void kernel_launch(void* const* d_in, const int* in_sizes, int n_in,
                              void* d_out, int out_size, void* d_ws, size_t ws_size,
                              hipStream_t stream) {
    (void)in_sizes; (void)n_in; (void)out_size; (void)ws_size;
    const float* x      = (const float*)d_in[0];
    const float* Wqkv_w = (const float*)d_in[1];
    const float* Wqkv_b = (const float*)d_in[2];
    const float* out_w  = (const float*)d_in[3];
    const float* out_b  = (const float*)d_in[4];
    float* out = (float*)d_out;

    const size_t nX  = (size_t)Bc * Sc * Ec;  // 8,388,608
    const size_t nWq = (size_t)3 * Ec * Ec;   // 12,582,912
    const size_t nWo = (size_t)Ec * Ec;       // 4,194,304

    bf16_t* xb  = (bf16_t*)d_ws;
    bf16_t* wqb = xb + nX;
    bf16_t* wob = wqb + nWq;
    bf16_t* qb  = wob + nWo;
    bf16_t* kb  = qb + nX;
    bf16_t* vb  = kb + nX;
    bf16_t* ctx = vb + nX;

    cvt_f32_bf16<<<(int)(nX / 4 / 256), 256, 0, stream>>>(x, xb, (int)(nX / 4));
    cvt_f32_bf16<<<(int)(nWq / 4 / 256), 256, 0, stream>>>(Wqkv_w, wqb, (int)(nWq / 4));
    cvt_f32_bf16<<<(int)(nWo / 4 / 256), 256, 0, stream>>>(out_w, wob, (int)(nWo / 4));

    // QKV projection: M=B*S=4096, N=3E=6144, K=E=2048
    gemm_wmma<true><<<dim3(3 * Ec / 128, (Bc * Sc) / 128), 256, 0, stream>>>(
        xb, wqb, Wqkv_b, qb, kb, vb, nullptr, Bc * Sc, 3 * Ec, Ec);

    // Attention
    attn_flash<<<dim3(Sc / 128, Bc * Hc), 256, 0, stream>>>(qb, kb, vb, ctx);

    // Output projection: M=4096, N=E, K=E, fp32 out + bias
    gemm_wmma<false><<<dim3(Ec / 128, (Bc * Sc) / 128), 256, 0, stream>>>(
        ctx, wob, out_b, nullptr, nullptr, nullptr, out, Bc * Sc, Ec, Ec);
}
